// LTConv_71511205479061
// MI455X (gfx1250) — matrix operand order repaired and checked by reading, not compiled, not run on hardware
//
#include <hip/hip_runtime.h>
#include <hip/hip_bf16.h>

typedef __attribute__((ext_vector_type(16))) __bf16 v16bf;
typedef __attribute__((ext_vector_type(8)))  float  v8f;

#define NNODES  100000
#define NEDGES  1600000
#define CH      64
#define CH2     128
#define NLAYERS 3
#define STRIPS  (NNODES / 16)   // 6250, exact

// ---------------------------------------------------------------- utilities
__global__ __launch_bounds__(256) void zero_f32_kernel(float* __restrict__ p, int n) {
    int i = blockIdx.x * blockDim.x + threadIdx.x;
    if (i < n) p[i] = 0.0f;
}

// Wt[l][n][k] = Ws[l][k][n]  (bf16, fragment-friendly transposed layout)
__global__ __launch_bounds__(256) void convert_w_kernel(const float* __restrict__ Ws,
                                                        __bf16* __restrict__ Wt) {
    int idx = blockIdx.x * blockDim.x + threadIdx.x;
    if (idx >= NLAYERS * CH * CH2) return;
    int l   = idx / (CH * CH2);
    int rem = idx % (CH * CH2);
    int n   = rem / CH;
    int k   = rem % CH;
    Wt[idx] = (__bf16)Ws[l * CH * CH2 + k * CH2 + n];
}

__global__ __launch_bounds__(256) void degree_kernel(const long long* __restrict__ ei,
                                                     float* __restrict__ deg) {
    int e = blockIdx.x * blockDim.x + threadIdx.x;
    if (e >= NEDGES) return;
    int d = (int)ei[NEDGES + e];
    __hip_atomic_fetch_add(&deg[d], 1.0f, __ATOMIC_RELAXED, __HIP_MEMORY_SCOPE_AGENT);
}

__global__ __launch_bounds__(256) void dinv_kernel(const float* __restrict__ deg,
                                                   float* __restrict__ dinv) {
    int i = blockIdx.x * blockDim.x + threadIdx.x;
    if (i < NNODES) dinv[i] = rsqrtf(deg[i] + 1.0f);   // self-loop adds 1
}

// ---------------------------------------------------------------- WMMA GEMM
// h[N,128] = x[N,64] @ W[64,128]; one wave per 16-row strip, 8 col tiles,
// K=64 split into two 16x16x32 bf16 WMMAs with f32 accumulation.
__global__ __launch_bounds__(256) void gemm_wmma_kernel(const float*  __restrict__ x,
                                                        const __bf16* __restrict__ Wt,
                                                        float* __restrict__ h) {
    int wid   = threadIdx.x >> 5;
    int lane  = threadIdx.x & 31;
    int strip = blockIdx.x * 8 + wid;          // wave-uniform
    if (strip >= STRIPS) return;               // whole wave exits together

    int mrow = strip * 16 + (lane & 15);
    int koff = (lane >> 4) << 3;               // 0 or 8 (ISA 16-bit A layout)
    const float* xr = x + mrow * CH;

    // A fragments for K-halves [0,32) and [32,64)
    v16bf a0, a1;
#pragma unroll
    for (int i = 0; i < 4; ++i) {
        int k = koff + 2 * i;
        a0[2*i]     = (__bf16)xr[k];        a0[2*i + 1] = (__bf16)xr[k + 1];
        a0[2*i + 8] = (__bf16)xr[16 + k];   a0[2*i + 9] = (__bf16)xr[16 + k + 1];
        a1[2*i]     = (__bf16)xr[32 + k];   a1[2*i + 1] = (__bf16)xr[32 + k + 1];
        a1[2*i + 8] = (__bf16)xr[48 + k];   a1[2*i + 9] = (__bf16)xr[48 + k + 1];
    }

    int nbase = lane & 15;
    int kb    = (lane >> 4) << 3;              // 0 or 8 (ISA 16-bit B layout)
    int rbase = strip * 16 + ((lane >> 4) << 3);

#pragma unroll
    for (int ct = 0; ct < 8; ++ct) {
        const __bf16* wr = Wt + (ct * 16 + nbase) * CH;   // Wt is [n][k]
        v16bf b0, b1;
#pragma unroll
        for (int i = 0; i < 8; ++i) {
            int kk = kb + 2*i + ((i >> 2) << 3);
            b0[2*i]     = wr[kk];      b0[2*i + 1] = wr[kk + 1];
            b1[2*i]     = wr[32 + kk]; b1[2*i + 1] = wr[32 + kk + 1];
        }
        v8f c = {};
        c = __builtin_amdgcn_wmma_f32_16x16x32_bf16(false, a0, false, b0,
                                                    (short)0, c, false, false);
        c = __builtin_amdgcn_wmma_f32_16x16x32_bf16(false, a1, false, b1,
                                                    (short)0, c, false, false);
        float* hw = h + ct * 16 + nbase;
#pragma unroll
        for (int r = 0; r < 8; ++r) hw[(rbase + r) * CH2] = c[r];
    }
}

// ---------------------------------------------------------------- layer ops
// agg = h * dinv[i]^2 + b   (self-loop term + bias; also zero-initializes agg)
__global__ __launch_bounds__(256) void init_agg_kernel(const float* __restrict__ h,
                                                       const float* __restrict__ dinv,
                                                       const float* __restrict__ b,
                                                       float* __restrict__ agg) {
    int idx = blockIdx.x * blockDim.x + threadIdx.x;
    if (idx >= NNODES * CH2) return;
    int i = idx >> 7, c = idx & 127;
    float di = dinv[i];
    agg[idx] = h[idx] * di * di + b[c];
}

// 32 lanes per edge, float4 per lane: one 512B gather line + one 512B atomic line
__global__ __launch_bounds__(256) void scatter_kernel(const long long* __restrict__ ei,
                                                      const float* __restrict__ h,
                                                      const float* __restrict__ dinv,
                                                      float* __restrict__ agg) {
    long long gid = (long long)blockIdx.x * blockDim.x + threadIdx.x;
    int e    = (int)(gid >> 5);
    int lane = (int)(gid & 31);
    if (e >= NEDGES) return;
    int s = (int)ei[e];
    int d = (int)ei[NEDGES + e];
    float coef = dinv[s] * dinv[d];
    const float4 hv = *(const float4*)(h + s * CH2 + lane * 4);
    float* ap = agg + d * CH2 + lane * 4;
    __hip_atomic_fetch_add(ap + 0, hv.x * coef, __ATOMIC_RELAXED, __HIP_MEMORY_SCOPE_AGENT);
    __hip_atomic_fetch_add(ap + 1, hv.y * coef, __ATOMIC_RELAXED, __HIP_MEMORY_SCOPE_AGENT);
    __hip_atomic_fetch_add(ap + 2, hv.z * coef, __ATOMIC_RELAXED, __HIP_MEMORY_SCOPE_AGENT);
    __hip_atomic_fetch_add(ap + 3, hv.w * coef, __ATOMIC_RELAXED, __HIP_MEMORY_SCOPE_AGENT);
}

// x_out = agg[:, :64] * sigmoid(agg[:, 64:]) + x_in
__global__ __launch_bounds__(256) void glu_kernel(const float* __restrict__ agg,
                                                  const float* __restrict__ xin,
                                                  float* __restrict__ xout) {
    int idx = blockIdx.x * blockDim.x + threadIdx.x;
    if (idx >= NNODES * CH) return;
    int i = idx >> 6, c = idx & 63;
    float a = agg[i * CH2 + c];
    float g = agg[i * CH2 + 64 + c];
    float sg = 1.0f / (1.0f + __expf(-g));
    xout[idx] = a * sg + xin[idx];
}

// ---------------------------------------------------------------- launcher
extern "C" void kernel_launch(void* const* d_in, const int* in_sizes, int n_in,
                              void* d_out, int out_size, void* d_ws, size_t ws_size,
                              hipStream_t stream) {
    const float*     x  = (const float*)d_in[0];
    const float*     Ws = (const float*)d_in[1];
    const float*     bs = (const float*)d_in[2];
    const long long* ei = (const long long*)d_in[3];
    float* out = (float*)d_out;

    char*  ws  = (char*)d_ws;
    size_t off = 0;
    auto alloc = [&](size_t bytes) -> void* {
        void* p = ws + off;
        off = (off + bytes + 255) & ~(size_t)255;
        return p;
    };
    float*  h    = (float*)alloc(sizeof(float) * NNODES * CH2);   // 51.2 MB (L2-resident)
    float*  agg  = (float*)alloc(sizeof(float) * NNODES * CH2);   // 51.2 MB (L2-resident)
    float*  x1   = (float*)alloc(sizeof(float) * NNODES * CH);
    float*  x2   = (float*)alloc(sizeof(float) * NNODES * CH);
    float*  deg  = (float*)alloc(sizeof(float) * NNODES);
    float*  dinv = (float*)alloc(sizeof(float) * NNODES);
    __bf16* Wt   = (__bf16*)alloc(sizeof(__bf16) * NLAYERS * CH * CH2);

    // one-time per call: degrees + normalization + bf16 weights
    zero_f32_kernel<<<(NNODES + 255) / 256, 256, 0, stream>>>(deg, NNODES);
    convert_w_kernel<<<(NLAYERS * CH * CH2 + 255) / 256, 256, 0, stream>>>(Ws, Wt);
    degree_kernel<<<(NEDGES + 255) / 256, 256, 0, stream>>>(ei, deg);
    dinv_kernel<<<(NNODES + 255) / 256, 256, 0, stream>>>(deg, dinv);

    const float* xin = x;
    float* xouts[NLAYERS] = { x1, x2, out };
    for (int l = 0; l < NLAYERS; ++l) {
        gemm_wmma_kernel<<<(STRIPS + 7) / 8, 256, 0, stream>>>(xin, Wt + l * CH * CH2, h);
        init_agg_kernel<<<(NNODES * CH2 + 255) / 256, 256, 0, stream>>>(h, dinv,
                                                                        bs + l * CH2, agg);
        scatter_kernel<<<(int)(((long long)NEDGES * 32 + 255) / 256), 256, 0, stream>>>(
            ei, h, dinv, agg);
        glu_kernel<<<(NNODES * CH + 255) / 256, 256, 0, stream>>>(agg, xin, xouts[l]);
        xin = xouts[l];
    }
}
